// DragnnLSTM_49512382988779
// MI455X (gfx1250) — matrix operand order repaired and checked by reading, not compiled, not run on hardware
//
#include <hip/hip_runtime.h>

// ---------------------------------------------------------------------------
// DRAGNN-LSTM for MI455X (gfx1250, wave32)
//
// Phase 1: xi/xo/xc = X @ {x2i,x2o,x2c} -> bf16 WMMA GEMM, 128x64 WG tile,
//          double-buffered LDS, A staged via GLOBAL_LOAD_ASYNC_TO_LDS_B128
//          (ASYNCcnt) when the toolchain exposes the builtin.
// Phase 2: sequential scan; recurrent weights transposed to bf16 (42 MB ->
//          resident in the 192 MB L2); persistent 64-WG grid with
//          device-scope sense-reversal barrier between dependency phases.
// ---------------------------------------------------------------------------

#define LSEQ 4096
#define DIN  1024
#define HID  2048

typedef __bf16  v16bf __attribute__((ext_vector_type(16)));
typedef float   v8f   __attribute__((ext_vector_type(8)));
typedef int     v4i_vs __attribute__((vector_size(16)));   // builtin's pointee

#if __has_builtin(__builtin_amdgcn_global_load_async_to_lds_b128) && \
    __has_builtin(__builtin_amdgcn_s_wait_asynccnt)
#define HAVE_ASYNC_LDS 1
#else
#define HAVE_ASYNC_LDS 0
#endif

#define AS1 __attribute__((address_space(1)))
#define AS3 __attribute__((address_space(3)))

#if HAVE_ASYNC_LDS
// Generic->AS3: low 32 bits of a generic LDS pointer are the LDS offset
// (ISA 10.2); generic->AS1 is value-preserving.
// Builtin wants non-const v4i pointers (per clang diagnostic in round 2).
__device__ __forceinline__ AS3 v4i_vs* to_lds(void* p) {
  return (AS3 v4i_vs*)(__UINTPTR_TYPE__)p;
}
__device__ __forceinline__ AS1 v4i_vs* to_glb(const void* p) {
  return (AS1 v4i_vs*)(__UINTPTR_TYPE__)p;
}
#endif

// ---- workspace layout (bytes) ---------------------------------------------
#define OFF_XBF    ((size_t)0)                         // 4096*1024 u16 = 8 MB
#define OFF_WXBF   ((size_t)8388608)                   // 3*1024*2048 u16 = 12 MB
#define OFF_XPROJ  ((size_t)20971520)                  // 3*4096*2048 f32 = 96 MB
#define OFF_WT     ((size_t)121634816)                 // 5*2048*2048 u16 = 40 MB
#define OFF_SYNC   ((size_t)163577856)                 // sync counters

// ---------------------------------------------------------------------------
__device__ __forceinline__ unsigned short f2bf(float f) {
  union { float f; unsigned u; } c; c.f = f;
  unsigned u = c.u;
  unsigned r = (u + 0x7fffu + ((u >> 16) & 1u)) >> 16;   // round-nearest-even
  return (unsigned short)r;
}

__global__ __launch_bounds__(256) void convert_f32_bf16(
    const float* __restrict__ src, unsigned short* __restrict__ dst, int n) {
  for (int i = blockIdx.x * blockDim.x + threadIdx.x; i < n;
       i += gridDim.x * blockDim.x)
    dst[i] = f2bf(src[i]);
}

// WT[j][k] = W[k][j], f32 -> bf16, LDS 32x32 tile transpose for coalescing
__global__ __launch_bounds__(256) void transpose_f32_bf16(
    const float* __restrict__ W, unsigned short* __restrict__ WT) {
  __shared__ float tile[32][33];
  const int bx = blockIdx.x * 32, by = blockIdx.y * 32;
  const int tx = threadIdx.x, ty = threadIdx.y;     // block (32, 8)
  for (int i = ty; i < 32; i += 8)
    tile[i][tx] = W[(size_t)(by + i) * HID + bx + tx];
  __syncthreads();
  for (int i = ty; i < 32; i += 8)
    WT[(size_t)(bx + i) * HID + by + tx] = f2bf(tile[tx][i]);
}

// ---------------------------------------------------------------------------
// WMMA GEMM:  Out[z] (4096x2048 f32) = Xbf (4096x1024 bf16) @ Wbf[z] (1024x2048)
// 256 threads = 8 waves; WG tile 128x64; each wave owns a 32x32 region
// (2 A-frags x 2 B-frags -> 4 WMMAs per 32-deep K step).  LDS pitch = 20
// dwords (80 B): 16-byte aligned for async B128 writes, and m*20 mod 64 is
// distinct for 16 lanes -> conflict-free fragment gathers.
// ---------------------------------------------------------------------------
__device__ __forceinline__ void gemm_storeB(unsigned (*Bb)[20], int kk, int np,
                                            uint4 v) {
  unsigned short* BtH = (unsigned short*)&Bb[0][0];
  unsigned e[4] = {v.x, v.y, v.z, v.w};
#pragma unroll
  for (int d = 0; d < 4; ++d) {
    BtH[(np + 2 * d + 0) * 40 + kk] = (unsigned short)(e[d] & 0xffffu);
    BtH[(np + 2 * d + 1) * 40 + kk] = (unsigned short)(e[d] >> 16);
  }
}

__global__ __launch_bounds__(256) void xproj_gemm_wmma(
    const unsigned short* __restrict__ Xbf,
    const unsigned short* __restrict__ Wbf,
    float* __restrict__ Out) {
  const unsigned short* Wm = Wbf + (size_t)blockIdx.z * DIN * HID;
  float*                O  = Out + (size_t)blockIdx.z * LSEQ * HID;
  const int m0 = blockIdx.y * 128;
  const int n0 = blockIdx.x * 64;

  __shared__ unsigned Abuf[2][128][20];   // 128 rows x 32 bf16 (+pad)
  __shared__ unsigned Bbuf[2][64][20];    // 64 cols x 32 bf16, transposed

  const int tid  = threadIdx.x;
  const int wave = tid >> 5;
  const int lane = tid & 31;
  const int rw = (wave & 3) * 32;        // wave's 32-row region
  const int cw = (wave >> 2) * 32;       // wave's 32-col region

  // A staging: two 16 B chunks/thread (rows ar and ar+64)
  const int ar = tid >> 2;               // 0..63
  const int ap = tid & 3;                // 16 B part within 64 B row
  // B staging: one b128/thread, transposed into LDS
  const int kk = tid >> 3;               // 0..31
  const int np = (tid & 7) * 8;          // 0..56

  v8f a00 = {}, a01 = {}, a10 = {}, a11 = {};

  // ---- prologue: stage K-tile 0 into buffer 0 ----
  {
#if HAVE_ASYNC_LDS
    __builtin_amdgcn_global_load_async_to_lds_b128(
        to_glb(Xbf + (size_t)(m0 + ar) * DIN + ap * 8),
        to_lds(&Abuf[0][ar][ap * 4]), 0, 0);
    __builtin_amdgcn_global_load_async_to_lds_b128(
        to_glb(Xbf + (size_t)(m0 + ar + 64) * DIN + ap * 8),
        to_lds(&Abuf[0][ar + 64][ap * 4]), 0, 0);
#else
    *(uint4*)&Abuf[0][ar][ap * 4] =
        *(const uint4*)(Xbf + (size_t)(m0 + ar) * DIN + ap * 8);
    *(uint4*)&Abuf[0][ar + 64][ap * 4] =
        *(const uint4*)(Xbf + (size_t)(m0 + ar + 64) * DIN + ap * 8);
#endif
    gemm_storeB(&Bbuf[0][0], kk, np,
                *(const uint4*)(Wm + (size_t)kk * HID + n0 + np));
#if HAVE_ASYNC_LDS
    __builtin_amdgcn_s_wait_asynccnt(0);
#endif
    __syncthreads();
  }

  for (int kt = 0; kt < DIN / 32; ++kt) {
    const int cur = kt & 1, nxt = cur ^ 1;
    const int k1 = (kt + 1) * 32;
    const bool havenext = (kt + 1 < DIN / 32);

    uint4 bv;
#if !HAVE_ASYNC_LDS
    uint4 av0, av1;
#endif
    if (havenext) {
#if HAVE_ASYNC_LDS
      // DMA next A tile straight into LDS; overlaps with the WMMAs below.
      __builtin_amdgcn_global_load_async_to_lds_b128(
          to_glb(Xbf + (size_t)(m0 + ar) * DIN + k1 + ap * 8),
          to_lds(&Abuf[nxt][ar][ap * 4]), 0, 0);
      __builtin_amdgcn_global_load_async_to_lds_b128(
          to_glb(Xbf + (size_t)(m0 + ar + 64) * DIN + k1 + ap * 8),
          to_lds(&Abuf[nxt][ar + 64][ap * 4]), 0, 0);
#else
      av0 = *(const uint4*)(Xbf + (size_t)(m0 + ar) * DIN + k1 + ap * 8);
      av1 = *(const uint4*)(Xbf + (size_t)(m0 + ar + 64) * DIN + k1 + ap * 8);
#endif
      bv = *(const uint4*)(Wm + (size_t)(k1 + kk) * HID + n0 + np);
    }

    // ---- fragments from LDS (contiguous k order: k = 16*half-wave + s) + 4 WMMAs ----
    {
      union { v16bf v; unsigned u[8]; } fa0, fa1, fb0, fb1;
      const int g4 = (lane >> 4) * 4;            // A: hi half-wave -> K+16
      const int g8 = (lane >> 4) * 8;            // B: hi half-wave -> K+16
      const int mr = rw + (lane & 15);
      const int nc = cw + (lane & 15);
#pragma unroll
      for (int p = 0; p < 8; ++p) {
        const int di = p + g8;
        fa0.u[p] = Abuf[cur][mr][di];
        fa1.u[p] = Abuf[cur][mr + 16][di];
        fb0.u[p] = Bbuf[cur][nc][p + g8];
        fb1.u[p] = Bbuf[cur][nc + 16][p + g8];
      }
      a00 = __builtin_amdgcn_wmma_f32_16x16x32_bf16(
          false, fa0.v, false, fb0.v, (short)0, a00, false, false);
      a01 = __builtin_amdgcn_wmma_f32_16x16x32_bf16(
          false, fa0.v, false, fb1.v, (short)0, a01, false, false);
      a10 = __builtin_amdgcn_wmma_f32_16x16x32_bf16(
          false, fa1.v, false, fb0.v, (short)0, a10, false, false);
      a11 = __builtin_amdgcn_wmma_f32_16x16x32_bf16(
          false, fa1.v, false, fb1.v, (short)0, a11, false, false);
    }

    if (havenext) {
#if !HAVE_ASYNC_LDS
      *(uint4*)&Abuf[nxt][ar][ap * 4]      = av0;
      *(uint4*)&Abuf[nxt][ar + 64][ap * 4] = av1;
#endif
      gemm_storeB(&Bbuf[nxt][0], kk, np, bv);
#if HAVE_ASYNC_LDS
      __builtin_amdgcn_s_wait_asynccnt(0);   // next A tile landed in LDS
#endif
      __syncthreads();                       // cur reads done, nxt visible
    }
  }

  // ---- store C: VGPR r -> M = r (+8 hi half-wave), N = lane&15 ----
  const int chi = (lane >> 4) ? 8 : 0;
  const int ncl = lane & 15;
  const int bm = m0 + rw, bn = n0 + cw;
#pragma unroll
  for (int r = 0; r < 8; ++r) {
    O[(size_t)(bm + chi + r) * HID + bn + ncl]           = a00[r];
    O[(size_t)(bm + chi + r) * HID + bn + 16 + ncl]      = a01[r];
    O[(size_t)(bm + 16 + chi + r) * HID + bn + ncl]      = a10[r];
    O[(size_t)(bm + 16 + chi + r) * HID + bn + 16 + ncl] = a11[r];
  }
}

// ---------------------------------------------------------------------------
// Persistent-grid scan kernel
// ---------------------------------------------------------------------------
__global__ void init_sync(unsigned* s) {
  if (threadIdx.x < 8) s[threadIdx.x] = 0u;
}

__device__ __forceinline__ void grid_sync(unsigned* cnt, unsigned* gen,
                                          unsigned nwg) {
  __threadfence();                        // release prior global writes
  __syncthreads();
  if (threadIdx.x == 0) {
    unsigned g = __hip_atomic_load(gen, __ATOMIC_RELAXED,
                                   __HIP_MEMORY_SCOPE_AGENT);
    unsigned old = __hip_atomic_fetch_add(cnt, 1u, __ATOMIC_ACQ_REL,
                                          __HIP_MEMORY_SCOPE_AGENT);
    if (old == nwg - 1) {
      __hip_atomic_store(cnt, 0u, __ATOMIC_RELAXED, __HIP_MEMORY_SCOPE_AGENT);
      __hip_atomic_fetch_add(gen, 1u, __ATOMIC_ACQ_REL,
                             __HIP_MEMORY_SCOPE_AGENT);
    } else {
      while (__hip_atomic_load(gen, __ATOMIC_ACQUIRE,
                               __HIP_MEMORY_SCOPE_AGENT) == g)
        __builtin_amdgcn_s_sleep(1);
    }
  }
  __syncthreads();
  __threadfence();                        // acquire: drop stale L0/WGP$ lines
}

// 2048-length dot of bf16 weight row (transposed, contiguous, L2-resident)
// vs LDS f32 vector; 8 threads/column, interleaved 16 B chunks so a wave's
// LDS reads land in distinct bank groups.
__device__ __forceinline__ float dot2048(const uint4* __restrict__ w,
                                         const float* __restrict__ x, int r) {
  float s = 0.f;
#pragma unroll 4
  for (int c = r; c < 256; c += 8) {
    uint4 pk = w[c];
    const float* xp = x + c * 8;
    s = fmaf(__uint_as_float(pk.x << 16),         xp[0], s);
    s = fmaf(__uint_as_float(pk.x & 0xffff0000u), xp[1], s);
    s = fmaf(__uint_as_float(pk.y << 16),         xp[2], s);
    s = fmaf(__uint_as_float(pk.y & 0xffff0000u), xp[3], s);
    s = fmaf(__uint_as_float(pk.z << 16),         xp[4], s);
    s = fmaf(__uint_as_float(pk.z & 0xffff0000u), xp[5], s);
    s = fmaf(__uint_as_float(pk.w << 16),         xp[6], s);
    s = fmaf(__uint_as_float(pk.w & 0xffff0000u), xp[7], s);
  }
  return s;
}

#define SCAN_WGS 64

__global__ __launch_bounds__(256) void lstm_scan_kernel(
    const float* __restrict__ xi, const float* __restrict__ xo,
    const float* __restrict__ xc,
    const unsigned short* __restrict__ h2iT,
    const unsigned short* __restrict__ c2iT,
    const unsigned short* __restrict__ h2cT,
    const unsigned short* __restrict__ h2oT,
    const unsigned short* __restrict__ c2oT,
    const float* __restrict__ bi, const float* __restrict__ bo,
    const float* __restrict__ bc,
    float* __restrict__ hidden, float* __restrict__ cell,
    unsigned* __restrict__ sync) {
  __shared__ float hs[HID];
  __shared__ float cs[HID];

  const int tid = threadIdx.x;
  const int col = blockIdx.x * 32 + (tid >> 3);   // global column j
  const int r   = tid & 7;                        // K-split index

  const float bij = bi[col], boj = bo[col], bcj = bc[col];

  const uint4* wh2i = (const uint4*)(h2iT + (size_t)col * HID);
  const uint4* wc2i = (const uint4*)(c2iT + (size_t)col * HID);
  const uint4* wh2c = (const uint4*)(h2cT + (size_t)col * HID);
  const uint4* wh2o = (const uint4*)(h2oT + (size_t)col * HID);
  const uint4* wc2o = (const uint4*)(c2oT + (size_t)col * HID);

  float ctj = 0.f;   // valid on r==0 lanes only

  for (int t = 0; t < LSEQ; ++t) {
    // ---- stage h_{t-1}, c_{t-1} into LDS ----
    if (t == 0) {
      for (int k = tid; k < HID; k += 256) { hs[k] = 0.f; cs[k] = 0.f; }
    } else {
      const float* hp = hidden + (size_t)(t - 1) * HID;
      const float* cp = cell   + (size_t)(t - 1) * HID;
      for (int k = tid; k < HID; k += 256) { hs[k] = hp[k]; cs[k] = cp[k]; }
    }
    __syncthreads();

    // ---- phase A: i_pre = h.h2i + c.c2i ; w_pre = h.h2c ; oh = h.h2o ----
    float pi = dot2048(wh2i, hs, r) + dot2048(wc2i, cs, r);
    float pw = dot2048(wh2c, hs, r);
    float po = dot2048(wh2o, hs, r);
#pragma unroll
    for (int off = 4; off > 0; off >>= 1) {       // 8-lane reduction
      pi += __shfl_down(pi, off, 8);
      pw += __shfl_down(pw, off, 8);
      po += __shfl_down(po, off, 8);
    }

    if (r == 0) {
      const float i_pre = xi[(size_t)t * HID + col] + bij + pi;
      const float it    = 1.f / (1.f + __expf(-i_pre));
      const float wt    = tanhf(xc[(size_t)t * HID + col] + bcj + pw);
      ctj = it * wt + (1.f - it) * cs[col];
      cell[(size_t)t * HID + col] = ctj;
    }
    grid_sync(sync, sync + 1, SCAN_WGS);

    // ---- phase B needs full c_t: reload into LDS ----
    {
      const float* cp = cell + (size_t)t * HID;
      for (int k = tid; k < HID; k += 256) cs[k] = cp[k];
    }
    __syncthreads();

    float pco = dot2048(wc2o, cs, r);
#pragma unroll
    for (int off = 4; off > 0; off >>= 1) pco += __shfl_down(pco, off, 8);

    if (r == 0) {
      const float o_pre = xo[(size_t)t * HID + col] + boj + po + pco;
      const float ot    = 1.f / (1.f + __expf(-o_pre));
      hidden[(size_t)t * HID + col] = ot * tanhf(ctj);
    }
    grid_sync(sync, sync + 1, SCAN_WGS);  // h_t visible before next step reads
  }
}

// ---------------------------------------------------------------------------
extern "C" void kernel_launch(void* const* d_in, const int* in_sizes, int n_in,
                              void* d_out, int out_size, void* d_ws,
                              size_t ws_size, hipStream_t stream) {
  (void)in_sizes; (void)n_in; (void)out_size; (void)ws_size;

  const float* X   = (const float*)d_in[0];
  const float* x2i = (const float*)d_in[1];
  const float* h2i = (const float*)d_in[2];
  const float* c2i = (const float*)d_in[3];
  const float* bi  = (const float*)d_in[4];
  const float* x2o = (const float*)d_in[5];
  const float* h2o = (const float*)d_in[6];
  const float* c2o = (const float*)d_in[7];
  const float* bo  = (const float*)d_in[8];
  const float* x2c = (const float*)d_in[9];
  const float* h2c = (const float*)d_in[10];
  const float* bc  = (const float*)d_in[11];

  char* ws = (char*)d_ws;
  unsigned short* xbf   = (unsigned short*)(ws + OFF_XBF);
  unsigned short* wxbf  = (unsigned short*)(ws + OFF_WXBF);
  float*          xproj = (float*)(ws + OFF_XPROJ);
  unsigned short* wt    = (unsigned short*)(ws + OFF_WT);
  unsigned*       sync  = (unsigned*)(ws + OFF_SYNC);

  const size_t WN = (size_t)DIN * HID;
  const size_t HH = (size_t)HID * HID;
  const size_t LH = (size_t)LSEQ * HID;

  // ---- fp32 -> bf16 conversions ----
  convert_f32_bf16<<<2048, 256, 0, stream>>>(X, xbf, LSEQ * DIN);
  convert_f32_bf16<<<1024, 256, 0, stream>>>(x2i, wxbf + 0 * WN, (int)WN);
  convert_f32_bf16<<<1024, 256, 0, stream>>>(x2o, wxbf + 1 * WN, (int)WN);
  convert_f32_bf16<<<1024, 256, 0, stream>>>(x2c, wxbf + 2 * WN, (int)WN);

  // ---- recurrent weights: transpose + bf16 (42 MB -> L2-resident) ----
  dim3 tg(HID / 32, HID / 32), tb(32, 8);
  transpose_f32_bf16<<<tg, tb, 0, stream>>>(h2i, wt + 0 * HH);
  transpose_f32_bf16<<<tg, tb, 0, stream>>>(c2i, wt + 1 * HH);
  transpose_f32_bf16<<<tg, tb, 0, stream>>>(h2c, wt + 2 * HH);
  transpose_f32_bf16<<<tg, tb, 0, stream>>>(h2o, wt + 3 * HH);
  transpose_f32_bf16<<<tg, tb, 0, stream>>>(c2o, wt + 4 * HH);

  // ---- xi/xo/xc via WMMA GEMM (z selects weight/output) ----
  xproj_gemm_wmma<<<dim3(HID / 64, LSEQ / 128, 3), 256, 0, stream>>>(
      xbf, wxbf, xproj);

  // ---- sequential scan ----
  init_sync<<<1, 32, 0, stream>>>(sync);
  float* hidden = (float*)d_out;
  float* cell   = hidden + LH;
  lstm_scan_kernel<<<SCAN_WGS, 256, 0, stream>>>(
      xproj, xproj + LH, xproj + 2 * LH,
      wt + 0 * HH, wt + 1 * HH, wt + 2 * HH, wt + 3 * HH, wt + 4 * HH,
      bi, bo, bc, hidden, cell, sync);
}
